// SegAttnModel_88811333746873
// MI455X (gfx1250) — matrix operand order repaired and checked by reading, not compiled, not run on hardware
//
#include <hip/hip_runtime.h>
#include <hip/hip_bf16.h>
#include <stdint.h>

typedef __attribute__((ext_vector_type(16))) _Float16 v16h;
typedef __attribute__((ext_vector_type(8)))  float    v8f;
typedef __attribute__((ext_vector_type(4)))  unsigned int u32x4;
typedef __attribute__((ext_vector_type(4)))  int      i32x4;
typedef __attribute__((ext_vector_type(8)))  int      i32x8;

#if defined(__has_builtin)
#if __has_builtin(__builtin_amdgcn_tensor_load_to_lds)
#define HAVE_TDM 1
#endif
#endif
#ifndef HAVE_TDM
#define HAVE_TDM 0
#endif

// ---------------------------------------------------------------------------
// WMMA helpers (CDNA5 gfx1250, wave32)
// ---------------------------------------------------------------------------
__device__ inline v8f wmma_f16(v16h a, v16h b, v8f c) {
  // v_wmma_f32_16x16x32_f16
  return __builtin_amdgcn_wmma_f32_16x16x32_f16(false, a, false, b, (short)0, c,
                                                false, false);
}

// A-matrix (16x32 f16) element->K mapping per ISA 7.12.2:
//   elements 2v,2v+1: lanes0-15 K = {0..7 then 16..23}, lanes16-31 +8
__device__ constexpr int a_kofs(int e, int lhalf) {
  int v = e >> 1, jj = e & 1;
  return ((v < 4) ? (2 * v + jj) : (16 + 2 * (v - 4) + jj)) + lhalf * 8;
}

__device__ inline float sigm(float x) { return 1.f / (1.f + __expf(-x)); }

// ---------------------------------------------------------------------------
// Tensor Data Mover: 2-D f16 tile -> LDS.  D# per ISA cdna5 ch.8.
// ---------------------------------------------------------------------------
#if HAVE_TDM
__device__ inline void tdm_load_2d_f16(uint32_t lds_byte_addr, const _Float16* src,
                                       uint32_t rows, uint32_t row_elems,
                                       uint32_t row_stride_elems) {
  union { u32x4 v; uint32_t u[4]; } g0;
  const uint64_t ga = (uint64_t)(uintptr_t)src;
  g0.u[0] = 1u;                                                 // count=1, user D#
  g0.u[1] = lds_byte_addr;                                      // lds_addr
  g0.u[2] = (uint32_t)ga;                                       // global_addr[31:0]
  g0.u[3] = (uint32_t)((ga >> 32) & 0x01FFFFFFu) | (2u << 30);  // addr[56:32]|type=2
  union { i32x8 v; uint32_t u[8]; } g1;
  const uint64_t td0 = row_elems;         // tensor_dim0
  const uint64_t td1 = rows;              // tensor_dim1
  const uint64_t s0 = row_stride_elems;   // tensor_dim0_stride
  g1.u[0] = (1u << 16);                                                   // data_size=2B
  g1.u[1] = (uint32_t)(td0 << 16);                                        // td0 lo16
  g1.u[2] = (uint32_t)((td0 >> 16) & 0xFFFFu) | ((uint32_t)(td1 & 0xFFFFu) << 16);
  g1.u[3] = (uint32_t)((td1 >> 16) & 0xFFFFu) | ((row_elems & 0xFFFFu) << 16);  // tile_dim0
  g1.u[4] = (rows & 0xFFFFu);                                             // tile_dim1
  g1.u[5] = (uint32_t)(s0 & 0xFFFFFFFFu);
  g1.u[6] = (uint32_t)((s0 >> 32) & 0xFFFFu);                             // dim1_stride=0
  g1.u[7] = 0;
  const i32x4 z4 = {0, 0, 0, 0};
#if __clang_major__ >= 23
  const i32x8 z8 = {0, 0, 0, 0, 0, 0, 0, 0};
  __builtin_amdgcn_tensor_load_to_lds(g0.v, g1.v, z4, z4, z8, 0);
#else
  __builtin_amdgcn_tensor_load_to_lds(g0.v, g1.v, z4, z4, 0);
#endif
}
#endif

// ---------------------------------------------------------------------------
// Prep kernels
// ---------------------------------------------------------------------------
__global__ void prep_conv_kernel(const float* __restrict__ cw, const float* __restrict__ cb,
                                 const float* __restrict__ bg, const float* __restrict__ bb,
                                 const float* __restrict__ bm, const float* __restrict__ bv,
                                 _Float16* __restrict__ w16, float* __restrict__ bias,
                                 int OC, int KK, int KP) {
  int idx = blockIdx.x * blockDim.x + threadIdx.x;
  if (idx >= OC * KP) return;
  int oc = idx / KP, k = idx % KP;
  float scale = bg[oc] * rsqrtf(bv[oc] + 1e-5f);
  w16[idx] = (k < KK) ? (_Float16)(cw[oc * KK + k] * scale) : (_Float16)0.0f;
  if (k == 0) bias[oc] = (cb[oc] - bm[oc]) * scale + bb[oc];
}

__global__ void cvt_f32_to_f16(const float* __restrict__ s, _Float16* __restrict__ d, int n) {
  int i = blockIdx.x * blockDim.x + threadIdx.x;
  if (i < n) d[i] = (_Float16)s[i];
}

// x (64,1,64,512) f32 -> padded f16 [64][68][516], interior at (+2,+2)
__global__ void cvt_x_pad(const float* __restrict__ src, _Float16* __restrict__ dst) {
  int i = blockIdx.x * blockDim.x + threadIdx.x;
  if (i >= 64 * 64 * 512) return;
  int x = i % 512;
  int t = i / 512;
  int y = t % 64;
  int b = t / 64;
  dst[((size_t)b * 68 + (y + 2)) * 516 + (x + 2)] = (_Float16)src[i];
}

// dst[Cc][R] (f16) = transpose of src[R][Cc] (f32)
__global__ void transpose_f16(const float* __restrict__ src, _Float16* __restrict__ dst,
                              int R, int Cc) {
  int i = blockIdx.x * blockDim.x + threadIdx.x;
  if (i >= R * Cc) return;
  int r = i / Cc, c = i % Cc;
  dst[(size_t)c * R + r] = (_Float16)src[(size_t)r * Cc + c];
}

// ---------------------------------------------------------------------------
// Conv (implicit GEMM) + BN(folded) + ReLU, WMMA f16->f32.
// act: zero-padded [B][IC][H+2P][W+2P] f16 (P = KS/2), interior at (+P,+P).
// wgt: [OC][KP] f16 im2col-major, staged block-wide into LDS via TDM.
// out: [B][OC][H][W] f16 (post-ReLU, unpadded).
// One wave per 16(OC)x16(pixel) tile; block = 4 waves sharing one weight tile.
// ---------------------------------------------------------------------------
template <int IC, int OC, int KS, int H, int W>
__global__ void __launch_bounds__(128)
conv_bn_relu_wmma(const _Float16* __restrict__ act, const _Float16* __restrict__ wgt,
                  const float* __restrict__ bias, _Float16* __restrict__ out) {
  constexpr int KK = IC * KS * KS;
  constexpr int KP = (KK + 31) & ~31;
  constexpr int PAD = KS / 2;
  constexpr int HP = H + 2 * PAD;
  constexpr int WP = W + 2 * PAD;
  constexpr int NPIX = H * W;
  extern __shared__ _Float16 wlds[];  // [16][KP]

  const int wave = threadIdx.x >> 5;
  const int lane = threadIdx.x & 31;
  const int lhalf = lane >> 4;
  const int lm = lane & 15;
  const int ocb = blockIdx.y * 16;
  const int b = blockIdx.z;

  // --- stage this block's weight tile (16 x KP f16) into LDS ---
#if HAVE_TDM
  if (threadIdx.x < 32) {  // wave 0 drives the TDM DMA
    tdm_load_2d_f16((uint32_t)(uintptr_t)wlds, wgt + (size_t)ocb * KP, 16, KP, KP);
    __builtin_amdgcn_s_wait_tensorcnt(0);
  }
#else
  for (int i = threadIdx.x; i < 16 * KP; i += 128) wlds[i] = wgt[(size_t)ocb * KP + i];
#endif
  __syncthreads();

  const int p0 = (blockIdx.x * 4 + wave) * 16;
  if (p0 >= NPIX) return;  // wave-uniform
  const int y0 = p0 / W;   // W multiple of 16 -> whole tile on one row
  const int x0 = p0 % W;
  const _Float16* actb = act + (size_t)b * IC * HP * WP;

  v8f acc = {};
  for (int kk = 0; kk < KP; kk += 32) {
    // A from LDS (shared weight tile): M = lane&15, K per a_kofs
    v16h a;
#pragma unroll
    for (int e = 0; e < 16; ++e) a[e] = wlds[lm * KP + kk + a_kofs(e, lhalf)];
    // B = im2col of padded input: K per a_kofs, N = p0+lm
    v16h bmat;
#pragma unroll
    for (int e = 0; e < 16; ++e) {
      const int k = kk + a_kofs(e, lhalf);
      if (KP == KK || k < KK) {
        const int ic = k / (KS * KS);
        const int r = k % (KS * KS);
        const int ky = r / KS;  // 0..KS-1 into padded frame
        const int kx = r % KS;
        bmat[e] = actb[((size_t)ic * HP + (y0 + ky)) * WP + (x0 + kx + lm)];
      } else {
        bmat[e] = (_Float16)0.0f;
      }
    }
    acc = wmma_f16(a, bmat, acc);
  }
  // C/D layout: N = lane&15, M = v + 8*lhalf
  const int p = p0 + lm;
#pragma unroll
  for (int v = 0; v < 8; ++v) {
    const int m = v + lhalf * 8;
    float o = acc[v] + bias[ocb + m];
    o = o > 0.f ? o : 0.f;
    out[((size_t)b * OC + (ocb + m)) * NPIX + p] = (_Float16)o;
  }
}

// ---------------------------------------------------------------------------
// MaxPool (PH x PW, stride = window) f16 -> padded f16 plane (pad OP, zeroed
// beforehand by the host launcher).
// ---------------------------------------------------------------------------
template <int C_, int H, int W, int PH, int PW, int OP>
__global__ void maxpool_kernel(const _Float16* __restrict__ in, _Float16* __restrict__ out) {
  constexpr int OH = H / PH, OW = W / PW;
  constexpr int OHP = OH + 2 * OP, OWP = OW + 2 * OP;
  size_t idx = (size_t)blockIdx.x * blockDim.x + threadIdx.x;
  const size_t total = (size_t)64 * C_ * OH * OW;
  if (idx >= total) return;
  int ox = idx % OW;
  size_t t = idx / OW;
  int oy = t % OH; t /= OH;
  int ch = t % C_;
  int b = t / C_;
  const _Float16* p = in + (((size_t)b * C_ + ch) * H + oy * PH) * W + (size_t)ox * PW;
  float m = -1e30f;
#pragma unroll
  for (int dy = 0; dy < PH; ++dy)
#pragma unroll
    for (int dx = 0; dx < PW; ++dx) {
      float v = (float)p[dy * W + dx];
      m = v > m ? v : m;
    }
  out[(((size_t)b * C_ + ch) * OHP + (oy + OP)) * OWP + (ox + OP)] = (_Float16)m;
}

// ---------------------------------------------------------------------------
// wfm[b][p][e] = sum_c wf[e][c]*fm[b][c][p] + bf[e]   (WMMA, M=e, N=p, K=c)
// fm16: [B][256][128] f16, wf16: [256][256] f16 -> wfm: [B][128][256] f32.
// grid(8 ptiles, 4, 64), block 128; TDM stages 64 wf rows (32 KB) per block.
// ---------------------------------------------------------------------------
__global__ void __launch_bounds__(128)
wfm_precompute(const _Float16* __restrict__ fm16, const _Float16* __restrict__ wf16,
               const float* __restrict__ bf, float* __restrict__ wfm) {
  extern __shared__ _Float16 alds[];  // [64][256]
  const int wave = threadIdx.x >> 5, lane = threadIdx.x & 31;
  const int lhalf = lane >> 4, lm = lane & 15;
  const int p0 = blockIdx.x * 16;
  const int eb0 = blockIdx.y * 64;
  const int b = blockIdx.z;

#if HAVE_TDM
  if (threadIdx.x < 32) {
    tdm_load_2d_f16((uint32_t)(uintptr_t)alds, wf16 + (size_t)eb0 * 256, 64, 256, 256);
    __builtin_amdgcn_s_wait_tensorcnt(0);
  }
#else
  for (int i = threadIdx.x; i < 64 * 256; i += 128) alds[i] = wf16[(size_t)eb0 * 256 + i];
#endif
  __syncthreads();

  const int erow = wave * 16;  // this wave's A rows within the staged tile
  v8f acc = {};
  for (int kc = 0; kc < 8; ++kc) {
    const int kk = kc * 32;
    v16h a, bmat;
#pragma unroll
    for (int e = 0; e < 16; ++e) a[e] = alds[(erow + lm) * 256 + kk + a_kofs(e, lhalf)];
    const int k = kk + lane;
    const _Float16* frow = fm16 + ((size_t)b * 256 + (k - lane)) * 128;
#pragma unroll
    for (int e = 0; e < 16; ++e) bmat[e] = frow[(size_t)a_kofs(e, lhalf) * 128 + p0 + lm];
    acc = wmma_f16(a, bmat, acc);
  }
#pragma unroll
  for (int v = 0; v < 8; ++v) {
    const int e = eb0 + erow + v + lhalf * 8;
    const int p = p0 + lm;
    wfm[((size_t)b * 128 + p) * 256 + e] = acc[v] + bf[e];
  }
}

// ---------------------------------------------------------------------------
// Fused 98-step attention-LSTM decoder.
// grid = 4 blocks x 256 threads; block bg handles batches [bg*16, bg*16+16).
// All recurrent state lives in LDS (~242 KB of the WGP's 320 KB).
// WMMA (M=16 batch) for q = h@whT and gates = xi@w_ihT + h@w_hhT.
// ---------------------------------------------------------------------------
__global__ void __launch_bounds__(256)
decoder_kernel(const float* __restrict__ emb, const _Float16* __restrict__ whT,
               const float* __restrict__ bh, const float* __restrict__ wfm,
               const _Float16* __restrict__ fm16, const _Float16* __restrict__ w_ihT,
               const float* __restrict__ b_ih, const _Float16* __restrict__ w_hhT,
               const float* __restrict__ b_hh, const float* __restrict__ w_fc,
               const float* __restrict__ b_fc, float* __restrict__ outL,
               float* __restrict__ outLP, float* __restrict__ outP) {
  extern __shared__ char smem[];
  float* h = (float*)smem;             // [16][512]
  float* cc = h + 16 * 512;            // [16][512]
  float* q = cc + 16 * 512;            // [16][256]
  float* sc = q + 16 * 256;            // [16][128] scores -> attn
  float* gates = sc + 16 * 128;        // [16][2048]
  float* logits = gates + 16 * 2048;   // [16][80]
  int* yy = (int*)(logits + 16 * 80);  // [16]
  _Float16* xi = (_Float16*)(yy + 16); // [16][512] f16

  const int tid = threadIdx.x;
  const int wave = tid >> 5, lane = tid & 31;
  const int lhalf = lane >> 4, lm = lane & 15;
  const int b0 = blockIdx.x * 16;

  for (int i = tid; i < 16 * 512; i += 256) { h[i] = 0.f; cc[i] = 0.f; }
  if (tid < 16) yy[tid] = 1;  // SOS
  __syncthreads();

  for (int t = 0; t < 98; ++t) {
    // -- embedding lookup into xi[:, 0:256]
    for (int i = tid; i < 16 * 256; i += 256) {
      const int b = i >> 8, e = i & 255;
      xi[b * 512 + e] = (_Float16)emb[(size_t)yy[b] * 256 + e];
    }
    __syncthreads();

    // -- q = h @ wh^T + bh  (WMMA: M=batch16, N=e, K=512)
    for (int nt = wave * 2; nt < wave * 2 + 2; ++nt) {
      const int eb = nt * 16;
      v8f acc = {};
      for (int kc = 0; kc < 16; ++kc) {
        const int kk = kc * 32;
        v16h a, bmat;
#pragma unroll
        for (int e = 0; e < 16; ++e) a[e] = (_Float16)h[lm * 512 + kk + a_kofs(e, lhalf)];
        const int k = kk + lane;
#pragma unroll
        for (int e = 0; e < 16; ++e) bmat[e] = whT[(size_t)(kk + a_kofs(e, lhalf)) * 256 + eb + lm];
        acc = wmma_f16(a, bmat, acc);
      }
#pragma unroll
      for (int v = 0; v < 8; ++v) {
        const int m = v + lhalf * 8;
        q[m * 256 + eb + lm] = acc[v] + bh[eb + lm];
      }
    }
    __syncthreads();

    // -- scores[b][p] = q[b] . wfm[b][p][:]
    for (int i = tid; i < 16 * 128; i += 256) {
      const int b = i >> 7, p = i & 127;
      const float* wrow = wfm + ((size_t)(b0 + b) * 128 + p) * 256;
      const float* qrow = q + b * 256;
      float s = 0.f;
      for (int e = 0; e < 256; ++e) s += qrow[e] * wrow[e];
      sc[i] = s;
    }
    __syncthreads();

    // -- softmax over 128 pixels (one thread per batch row)
    if (tid < 16) {
      float mx = -1e30f;
      for (int p = 0; p < 128; ++p) mx = fmaxf(mx, sc[tid * 128 + p]);
      float sum = 0.f;
      for (int p = 0; p < 128; ++p) {
        float e = __expf(sc[tid * 128 + p] - mx);
        sc[tid * 128 + p] = e;
        sum += e;
      }
      const float inv = 1.f / sum;
      for (int p = 0; p < 128; ++p) sc[tid * 128 + p] *= inv;
    }
    __syncthreads();

    // -- context z -> xi[:, 256:512]
    for (int i = tid; i < 16 * 256; i += 256) {
      const int b = i >> 8, ch = i & 255;
      const _Float16* frow = fm16 + ((size_t)(b0 + b) * 256 + ch) * 128;
      const float* arow = sc + b * 128;
      float s = 0.f;
      for (int p = 0; p < 128; ++p) s += (float)frow[p] * arow[p];
      xi[b * 512 + 256 + ch] = (_Float16)s;
    }
    __syncthreads();

    // -- gates = xi @ w_ih^T + h @ w_hh^T + b  (WMMA: M=16, N=2048, K=512 x2)
    for (int nt = wave; nt < 128; nt += 8) {
      const int gb = nt * 16;
      v8f acc = {};
      for (int kc = 0; kc < 16; ++kc) {
        const int kk = kc * 32;
        v16h a, bmat;
#pragma unroll
        for (int e = 0; e < 16; ++e) a[e] = xi[lm * 512 + kk + a_kofs(e, lhalf)];
        const int k = kk + lane;
#pragma unroll
        for (int e = 0; e < 16; ++e) bmat[e] = w_ihT[(size_t)(kk + a_kofs(e, lhalf)) * 2048 + gb + lm];
        if (kk + 32 < 512)
          __builtin_prefetch(&w_ihT[(size_t)(k + 32) * 2048 + gb], 0, 1);  // global_prefetch_b8
        acc = wmma_f16(a, bmat, acc);
      }
      for (int kc = 0; kc < 16; ++kc) {
        const int kk = kc * 32;
        v16h a, bmat;
#pragma unroll
        for (int e = 0; e < 16; ++e) a[e] = (_Float16)h[lm * 512 + kk + a_kofs(e, lhalf)];
        const int k = kk + lane;
#pragma unroll
        for (int e = 0; e < 16; ++e) bmat[e] = w_hhT[(size_t)(kk + a_kofs(e, lhalf)) * 2048 + gb + lm];
        acc = wmma_f16(a, bmat, acc);
      }
#pragma unroll
      for (int v = 0; v < 8; ++v) {
        const int m = v + lhalf * 8;
        const int g = gb + lm;
        gates[m * 2048 + g] = acc[v] + b_ih[g] + b_hh[g];
      }
    }
    __syncthreads();

    // -- LSTM pointwise (torch gate order i,f,g,o)
    for (int i = tid; i < 16 * 512; i += 256) {
      const int b = i >> 9, j = i & 511;
      const float gi = gates[b * 2048 + j];
      const float gf = gates[b * 2048 + 512 + j];
      const float gg = gates[b * 2048 + 1024 + j];
      const float go = gates[b * 2048 + 1536 + j];
      const float c2 = sigm(gf) * cc[i] + sigm(gi) * tanhf(gg);
      cc[i] = c2;
      h[i] = sigm(go) * tanhf(c2);
    }
    __syncthreads();

    // -- logits = h2 @ w_fc^T + b_fc  (V=80, VALU)
    for (int i = tid; i < 16 * 80; i += 256) {
      const int b = i / 80, v = i % 80;
      const float* wrow = w_fc + (size_t)v * 512;
      const float* hrow = h + b * 512;
      float s = b_fc[v];
      for (int k = 0; k < 512; ++k) s += hrow[k] * wrow[k];
      logits[i] = s;
    }
    __syncthreads();

    // -- log_softmax + argmax + outputs (L:[B,V,T], LP:[B,V,T], P:[B,T])
    if (tid < 16) {
      const int b = tid;
      float mx = -1e30f;
      int am = 0;
      for (int v = 0; v < 80; ++v) {
        const float l = logits[b * 80 + v];
        if (l > mx) { mx = l; am = v; }
      }
      float sum = 0.f;
      for (int v = 0; v < 80; ++v) sum += __expf(logits[b * 80 + v] - mx);
      const float lse = mx + __logf(sum);
      const size_t base = (size_t)(b0 + b) * 80 * 98;
      for (int v = 0; v < 80; ++v) {
        const float l = logits[b * 80 + v];
        outL[base + (size_t)v * 98 + t] = l;
        outLP[base + (size_t)v * 98 + t] = l - lse;
      }
      outP[(size_t)(b0 + b) * 98 + t] = (float)am;
      yy[b] = am;
    }
    __syncthreads();
  }
}

// ---------------------------------------------------------------------------
// Host launcher
// ---------------------------------------------------------------------------
extern "C" void kernel_launch(void* const* d_in, const int* in_sizes, int n_in,
                              void* d_out, int out_size, void* d_ws, size_t ws_size,
                              hipStream_t stream) {
  (void)in_sizes; (void)n_in; (void)out_size; (void)ws_size;
  const float* x = (const float*)d_in[0];
  const float *cw[6], *cb[6], *bg[6], *bb[6], *bm[6], *bv[6];
  for (int i = 0; i < 6; ++i) {
    cw[i] = (const float*)d_in[1 + 6 * i + 0];
    cb[i] = (const float*)d_in[1 + 6 * i + 1];
    bg[i] = (const float*)d_in[1 + 6 * i + 2];
    bb[i] = (const float*)d_in[1 + 6 * i + 3];
    bm[i] = (const float*)d_in[1 + 6 * i + 4];
    bv[i] = (const float*)d_in[1 + 6 * i + 5];
  }
  const float* emb = (const float*)d_in[37];
  const float* wf = (const float*)d_in[38];
  const float* bf = (const float*)d_in[39];
  const float* wh = (const float*)d_in[40];
  const float* bh = (const float*)d_in[41];
  const float* w_ih = (const float*)d_in[42];
  const float* b_ih = (const float*)d_in[43];
  const float* w_hh = (const float*)d_in[44];
  const float* b_hh = (const float*)d_in[45];
  const float* w_fc = (const float*)d_in[46];
  const float* b_fc = (const float*)d_in[47];

  static const int OCs[6] = {32, 64, 128, 128, 256, 256};
  static const int ICs[6] = {1, 32, 64, 128, 128, 256};
  static const int KSs[6] = {5, 5, 3, 3, 3, 3};

  // --- workspace carving ---
  char* ws = (char*)d_ws;
  size_t off = 0;
  auto alloc = [&](size_t bytes) -> void* {
    void* p = ws + off;
    off = (off + bytes + 255) & ~(size_t)255;
    return p;
  };
  _Float16* w16[6];
  float* bias[6];
  int KK[6], KP[6];
  for (int i = 0; i < 6; ++i) {
    KK[i] = ICs[i] * KSs[i] * KSs[i];
    KP[i] = (KK[i] + 31) & ~31;
    w16[i] = (_Float16*)alloc((size_t)OCs[i] * KP[i] * 2);
    bias[i] = (float*)alloc((size_t)OCs[i] * 4);
  }
  // padded activation planes (pads zero-filled each launch)
  const size_t xpadN = (size_t)64 * 1 * 68 * 516;          // P=2 for conv0
  const size_t p0N = (size_t)64 * 32 * 36 * 260;           // 32x256, P=2 for conv1
  const size_t p1N = (size_t)64 * 64 * 18 * 130;           // 16x128, P=1
  const size_t p2N = (size_t)64 * 128 * 10 * 130;          // 8x128,  P=1
  const size_t p3N = (size_t)64 * 128 * 6 * 130;           // 4x128,  P=1
  const size_t p4N = (size_t)64 * 256 * 4 * 130;           // 2x128,  P=1
  const size_t p5N = (size_t)64 * 256 * 1 * 128;           // 1x128,  unpadded (fm)
  _Float16* xpad = (_Float16*)alloc(xpadN * 2);
  _Float16* a0 = (_Float16*)alloc(p0N * 2);
  _Float16* a1 = (_Float16*)alloc(p1N * 2);
  _Float16* a2 = (_Float16*)alloc(p2N * 2);
  _Float16* a3 = (_Float16*)alloc(p3N * 2);
  _Float16* a4 = (_Float16*)alloc(p4N * 2);
  _Float16* fm16 = (_Float16*)alloc(p5N * 2);
  _Float16* convbuf = (_Float16*)alloc((size_t)64 * 32 * 64 * 512 * 2);  // largest conv out
  _Float16* wf16 = (_Float16*)alloc((size_t)256 * 256 * 2);
  _Float16* whT = (_Float16*)alloc((size_t)512 * 256 * 2);
  _Float16* w_ihT = (_Float16*)alloc((size_t)512 * 2048 * 2);
  _Float16* w_hhT = (_Float16*)alloc((size_t)512 * 2048 * 2);
  float* wfm = (float*)alloc((size_t)64 * 128 * 256 * 4);

  // --- zero the padded planes (graph-capturable async memsets) ---
  hipMemsetAsync(xpad, 0, xpadN * 2, stream);
  hipMemsetAsync(a0, 0, p0N * 2, stream);
  hipMemsetAsync(a1, 0, p1N * 2, stream);
  hipMemsetAsync(a2, 0, p2N * 2, stream);
  hipMemsetAsync(a3, 0, p3N * 2, stream);
  hipMemsetAsync(a4, 0, p4N * 2, stream);

  // --- prep ---
  for (int i = 0; i < 6; ++i) {
    const int n = OCs[i] * KP[i];
    prep_conv_kernel<<<(n + 255) / 256, 256, 0, stream>>>(
        cw[i], cb[i], bg[i], bb[i], bm[i], bv[i], w16[i], bias[i], OCs[i], KK[i], KP[i]);
  }
  cvt_x_pad<<<(64 * 64 * 512 + 255) / 256, 256, 0, stream>>>(x, xpad);
  cvt_f32_to_f16<<<(256 * 256 + 255) / 256, 256, 0, stream>>>(wf, wf16, 256 * 256);
  transpose_f16<<<(256 * 512 + 255) / 256, 256, 0, stream>>>(wh, whT, 256, 512);
  transpose_f16<<<(2048 * 512 + 255) / 256, 256, 0, stream>>>(w_ih, w_ihT, 2048, 512);
  transpose_f16<<<(2048 * 512 + 255) / 256, 256, 0, stream>>>(w_hh, w_hhT, 2048, 512);

  // --- backbone: conv(WMMA + TDM weight staging) -> maxpool (into padded) ---
  conv_bn_relu_wmma<1, 32, 5, 64, 512>
      <<<dim3(512, 2, 64), 128, 16 * KP[0] * 2, stream>>>(xpad, w16[0], bias[0], convbuf);
  maxpool_kernel<32, 64, 512, 2, 2, 2>
      <<<(64 * 32 * 32 * 256 + 255) / 256, 256, 0, stream>>>(convbuf, a0);

  conv_bn_relu_wmma<32, 64, 5, 32, 256>
      <<<dim3(128, 4, 64), 128, 16 * KP[1] * 2, stream>>>(a0, w16[1], bias[1], convbuf);
  maxpool_kernel<64, 32, 256, 2, 2, 1>
      <<<(64 * 64 * 16 * 128 + 255) / 256, 256, 0, stream>>>(convbuf, a1);

  conv_bn_relu_wmma<64, 128, 3, 16, 128>
      <<<dim3(32, 8, 64), 128, 16 * KP[2] * 2, stream>>>(a1, w16[2], bias[2], convbuf);
  maxpool_kernel<128, 16, 128, 2, 1, 1>
      <<<(64 * 128 * 8 * 128 + 255) / 256, 256, 0, stream>>>(convbuf, a2);

  conv_bn_relu_wmma<128, 128, 3, 8, 128>
      <<<dim3(16, 8, 64), 128, 16 * KP[3] * 2, stream>>>(a2, w16[3], bias[3], convbuf);
  maxpool_kernel<128, 8, 128, 2, 1, 1>
      <<<(64 * 128 * 4 * 128 + 255) / 256, 256, 0, stream>>>(convbuf, a3);

  conv_bn_relu_wmma<128, 256, 3, 4, 128>
      <<<dim3(8, 16, 64), 128, 16 * KP[4] * 2, stream>>>(a3, w16[4], bias[4], convbuf);
  maxpool_kernel<256, 4, 128, 2, 1, 1>
      <<<(64 * 256 * 2 * 128 + 255) / 256, 256, 0, stream>>>(convbuf, a4);

  conv_bn_relu_wmma<256, 256, 3, 2, 128>
      <<<dim3(4, 16, 64), 128, 16 * KP[5] * 2, stream>>>(a4, w16[5], bias[5], convbuf);
  maxpool_kernel<256, 2, 128, 2, 1, 0>
      <<<(64 * 256 * 1 * 128 + 255) / 256, 256, 0, stream>>>(convbuf, fm16);

  // --- attention precompute: wfm[b][p][e] (TDM-staged wf tile) ---
  wfm_precompute<<<dim3(8, 4, 64), 128, 64 * 256 * 2, stream>>>(fm16, wf16, bf, wfm);

  // --- fused 98-step decoder ---
  float* outL = (float*)d_out;
  float* outLP = outL + (size_t)64 * 80 * 98;
  float* outP = outLP + (size_t)64 * 80 * 98;
  const size_t SMEM = (size_t)(16 * 512 * 2 + 16 * 256 + 16 * 128 + 16 * 2048 + 16 * 80) * 4 +
                      16 * 4 + (size_t)16 * 512 * 2;  // ~242 KB (< 320 KB WGP LDS)
  decoder_kernel<<<4, 256, SMEM, stream>>>(emb, whT, bh, wfm, fm16, w_ihT, b_ih, w_hhT, b_hh,
                                           w_fc, b_fc, outL, outLP, outP);
}
